// HGRNBitAttention_41283225649669
// MI455X (gfx1250) — 8-bit matrix operand order repaired and checked by reading, compiled, not run on hardware
//
#include <hip/hip_runtime.h>
#include <cstdint>


#define B_   4
#define L_   2048
#define D_   2048
#define M_   (B_ * L_)
#define K_   D_
#define N_   D_
#define NC_  8
#define CL_  (L_ / NC_)

typedef int v8i __attribute__((ext_vector_type(8)));

__global__ __launch_bounds__(256) void k_wabs_partial(const float* __restrict__ w,
                                                      float* __restrict__ part) {
    __shared__ float red[256];
    const int t = threadIdx.x;
    const size_t base = (size_t)blockIdx.x * 16384 + t;
    float s = 0.f;
#pragma unroll 8
    for (int j = 0; j < 64; ++j) s += fabsf(w[base + (size_t)j * 256]);
    red[t] = s;
    __syncthreads();
    for (int off = 128; off > 0; off >>= 1) {
        if (t < off) red[t] += red[t + off];
        __syncthreads();
    }
    if (t == 0) part[blockIdx.x] = red[0];
}

__global__ void k_wscale(const float* __restrict__ part, float* __restrict__ s_w) {
    const int t = threadIdx.x;
    if (t < 4) {
        float s = 0.f;
        for (int j = 0; j < 256; ++j) s += part[t * 256 + j];
        const float mean = s / ((float)N_ * (float)K_);
        s_w[t] = 1.0f / fmaxf(mean, 1e-5f);
    }
}

__global__ __launch_bounds__(256) void k_wquant(const float* __restrict__ w,
                                                int8_t* __restrict__ wq,
                                                const float* __restrict__ s_elem) {
    const float s = *s_elem;
    const size_t idx = (size_t)blockIdx.x * 256 + threadIdx.x;
    float q = rintf(w[idx] * s);
    q = fminf(fmaxf(q, -1.f), 1.f);
    wq[idx] = (int8_t)q;
}

__global__ __launch_bounds__(256) void k_actquant(const float* __restrict__ x,
                                                  int8_t* __restrict__ xq,
                                                  float* __restrict__ s_x) {
    __shared__ float red[256];
    const int m = blockIdx.x, t = threadIdx.x;
    const float* row = x + (size_t)m * D_;
    float v[8];
    float amax = 0.f;
#pragma unroll
    for (int j = 0; j < 8; ++j) {
        v[j] = row[t + j * 256];
        amax = fmaxf(amax, fabsf(v[j]));
    }
    red[t] = amax;
    __syncthreads();
    for (int off = 128; off > 0; off >>= 1) {
        if (t < off) red[t] = fmaxf(red[t], red[t + off]);
        __syncthreads();
    }
    const float s = 127.f / fmaxf(red[0], 1e-5f);
#pragma unroll
    for (int j = 0; j < 8; ++j) {
        float q = rintf(v[j] * s);
        q = fminf(fmaxf(q, -128.f), 127.f);
        xq[(size_t)m * D_ + t + j * 256] = (int8_t)q;
    }
    if (t == 0) s_x[m] = s;
}

__global__ __launch_bounds__(256) void k_gemm_iu8(const int8_t* __restrict__ A,
                                                  const int8_t* __restrict__ W,
                                                  float* __restrict__ out,
                                                  const float* __restrict__ s_row,
                                                  const float* __restrict__ s_w_elem,
                                                  int mode) {
    const int tid  = threadIdx.x;
    const int lane = tid & 31;
    const int wv   = tid >> 5;
    const int wm   = wv & 1;
    const int wn   = wv >> 1;
    const int m_wave = blockIdx.x * 64 + wm * 32;
    const int n_wave = blockIdx.y * 128 + wn * 32;
    const int lrow = lane & 15;
    const int hi   = lane >> 4;

    v8i acc[2][2] = {};
    const float sw = *s_w_elem;

    for (int k = 0; k < K_; k += 64) {
        v8i a[2], b[2];
#pragma unroll
        for (int mt = 0; mt < 2; ++mt) {
            const int8_t* ap = A + (size_t)(m_wave + mt * 16 + lrow) * K_ + k + hi * 8;
            const uint2 p0 = *(const uint2*)(ap + 0);
            const uint2 p1 = *(const uint2*)(ap + 16);
            const uint2 p2 = *(const uint2*)(ap + 32);
            const uint2 p3 = *(const uint2*)(ap + 48);
            a[mt][0] = (int)p0.x; a[mt][1] = (int)p0.y;
            a[mt][2] = (int)p1.x; a[mt][3] = (int)p1.y;
            a[mt][4] = (int)p2.x; a[mt][5] = (int)p2.y;
            a[mt][6] = (int)p3.x; a[mt][7] = (int)p3.y;
            __builtin_prefetch(ap + 64, 0, 1);
        }
#pragma unroll
        for (int nt = 0; nt < 2; ++nt) {
            const int8_t* bp = W + (size_t)(n_wave + nt * 16 + lrow) * K_ + k + hi * 8;
            const uint2 q0 = *(const uint2*)(bp + 0);
            const uint2 q1 = *(const uint2*)(bp + 16);
            const uint2 q2 = *(const uint2*)(bp + 32);
            const uint2 q3 = *(const uint2*)(bp + 48);
            b[nt][0] = (int)q0.x; b[nt][1] = (int)q0.y;
            b[nt][2] = (int)q1.x; b[nt][3] = (int)q1.y;
            b[nt][4] = (int)q2.x; b[nt][5] = (int)q2.y;
            b[nt][6] = (int)q3.x; b[nt][7] = (int)q3.y;
            __builtin_prefetch(bp + 64, 0, 1);
        }
#pragma unroll
        for (int mt = 0; mt < 2; ++mt)
#pragma unroll
            for (int nt = 0; nt < 2; ++nt)
                acc[mt][nt] = __builtin_amdgcn_wmma_i32_16x16x64_iu8(
                     true, a[mt],  true, b[nt],
                    acc[mt][nt],  false,  false);
    }

#pragma unroll
    for (int mt = 0; mt < 2; ++mt) {
#pragma unroll
        for (int r = 0; r < 8; ++r) {
            const int m = m_wave + mt * 16 + hi * 8 + r;
            const float inv = 1.0f / (s_row[m] * sw);
#pragma unroll
            for (int nt = 0; nt < 2; ++nt) {
                const int n = n_wave + nt * 16 + lrow;
                float val = (float)acc[mt][nt][r] * inv;
                if (mode == 1) val = 1.0f / (1.0f + __expf(-val));
                out[(size_t)m * N_ + n] = val;
            }
        }
    }
}

__global__ __launch_bounds__(256) void k_scan_local(const float* __restrict__ f,
                                                    float* __restrict__ io,
                                                    float* __restrict__ P,
                                                    float* __restrict__ A) {
    const int t = blockIdx.x * 256 + threadIdx.x;
    const int d  = t & (D_ - 1);
    const int bc = t >> 11;
    const int chunk = bc & (NC_ - 1);
    const int b  = bc >> 3;
    const size_t base = ((size_t)b * L_ + (size_t)chunk * CL_) * D_ + d;
    float h = 0.f, p = 1.f;
    for (int l = 0; l < CL_; ++l) {
        const size_t idx = base + (size_t)l * D_;
        const float fv = f[idx];
        const float iv = io[idx];
        const float z  = 1.f - fv;
        const float ii = iv * (z / (1.f + __expf(-z)));
        h = fv * h + ii;
        p *= fv;
        io[idx] = h;
    }
    P[t] = p;
    A[t] = h;
}

__global__ __launch_bounds__(256) void k_scan_carry(const float* __restrict__ P,
                                                    const float* __restrict__ A,
                                                    float* __restrict__ carry) {
    const int t = blockIdx.x * 256 + threadIdx.x;
    const int b = t >> 11;
    const int d = t & (D_ - 1);
    float hend = 0.f;
#pragma unroll
    for (int c = 0; c < NC_; ++c) {
        const size_t idx = ((size_t)(b * NC_ + c) << 11) + d;
        carry[idx] = hend;
        hend = A[idx] + P[idx] * hend;
    }
}

__global__ __launch_bounds__(256) void k_scan_fix(const float* __restrict__ f,
                                                  float* __restrict__ io,
                                                  const float* __restrict__ carry) {
    const int t = blockIdx.x * 256 + threadIdx.x;
    const float cy = carry[t];
    if (cy == 0.f) return;
    const int d  = t & (D_ - 1);
    const int bc = t >> 11;
    const int chunk = bc & (NC_ - 1);
    const int b  = bc >> 3;
    const size_t base = ((size_t)b * L_ + (size_t)chunk * CL_) * D_ + d;
    float F = 1.f;
    for (int l = 0; l < CL_; ++l) {
        const size_t idx = base + (size_t)l * D_;
        F *= f[idx];
        io[idx] += F * cy;
    }
}

__global__ __launch_bounds__(256) void k_ygate(const float* __restrict__ g,
                                               const float* __restrict__ h,
                                               const float* __restrict__ gnw,
                                               int8_t* __restrict__ yq,
                                               float* __restrict__ s_y) {
    __shared__ float red[256];
    const int m = blockIdx.x, t = threadIdx.x;
    const float* grow = g + (size_t)m * D_;
    const float* hrow = h + (size_t)m * D_;

    float gv[8];
    float ss = 0.f;
#pragma unroll
    for (int j = 0; j < 8; ++j) {
        gv[j] = grow[t + j * 256];
        ss += gv[j] * gv[j];
    }
    red[t] = ss;
    __syncthreads();
    for (int off = 128; off > 0; off >>= 1) {
        if (t < off) red[t] += red[t + off];
        __syncthreads();
    }
    const float rms = rsqrtf(red[0] / (float)D_ + 1e-5f);
    __syncthreads();

    float yv[8];
    float amax = 0.f;
#pragma unroll
    for (int j = 0; j < 8; ++j) {
        const int d = t + j * 256;
        const float o  = hrow[d];
        const float sg = 1.f / (1.f + __expf(-o));
        const float y  = gv[j] * rms * gnw[d] * o * sg;
        yv[j] = y;
        amax = fmaxf(amax, fabsf(y));
    }
    red[t] = amax;
    __syncthreads();
    for (int off = 128; off > 0; off >>= 1) {
        if (t < off) red[t] = fmaxf(red[t], red[t + off]);
        __syncthreads();
    }
    const float s = 127.f / fmaxf(red[0], 1e-5f);
#pragma unroll
    for (int j = 0; j < 8; ++j) {
        float q = rintf(yv[j] * s);
        q = fminf(fmaxf(q, -128.f), 127.f);
        yq[(size_t)m * D_ + t + j * 256] = (int8_t)q;
    }
    if (t == 0) s_y[m] = s;
}

extern "C" void kernel_launch(void* const* d_in, const int* in_sizes, int n_in,
                              void* d_out, int out_size, void* d_ws, size_t ws_size,
                              hipStream_t stream) {
    (void)in_sizes; (void)n_in; (void)out_size; (void)ws_size;

    const float* x   = (const float*)d_in[0];
    const float* wts[4] = { (const float*)d_in[1],
                            (const float*)d_in[2],
                            (const float*)d_in[3],
                            (const float*)d_in[4] };
    const float* gnw = (const float*)d_in[5];
    float* out = (float*)d_out;

    uint8_t* ws = (uint8_t*)d_ws;
    int8_t* xq     = (int8_t*)(ws + 0);
    int8_t* wq     = (int8_t*)(ws + 16777216);
    float*  s_x    = (float*)(ws + 33554432);
    float*  s_y    = (float*)(ws + 33587200);
    float*  s_w    = (float*)(ws + 33619968);
    float*  wpart  = (float*)(ws + 33620224);
    int8_t* yq     = (int8_t*)(ws + 33624320);
    float*  i_buf  = (float*)(ws + 50401536);
    float*  f_buf  = (float*)(ws + 117510400);
    float*  Pbuf   = (float*)(ws + 184619264);
    float*  Abuf   = (float*)(ws + 184881408);
    float*  cybuf  = (float*)(ws + 185143552);

    for (int wi = 0; wi < 4; ++wi)
        k_wabs_partial<<<256, 256, 0, stream>>>(wts[wi], wpart + wi * 256);
    k_wscale<<<1, 32, 0, stream>>>(wpart, s_w);
    for (int wi = 0; wi < 4; ++wi)
        k_wquant<<<(N_ * K_) / 256, 256, 0, stream>>>(
            wts[wi], wq + (size_t)wi * N_ * K_, s_w + wi);

    k_actquant<<<M_, 256, 0, stream>>>(x, xq, s_x);

    dim3 ggrid(M_ / 64, N_ / 128);
    k_gemm_iu8<<<ggrid, 256, 0, stream>>>(xq, wq + 0 * (size_t)N_ * K_, i_buf, s_x, s_w + 0, 0);
    k_gemm_iu8<<<ggrid, 256, 0, stream>>>(xq, wq + 1 * (size_t)N_ * K_, f_buf, s_x, s_w + 1, 1);
    k_gemm_iu8<<<ggrid, 256, 0, stream>>>(xq, wq + 2 * (size_t)N_ * K_, out,   s_x, s_w + 2, 0);

    k_scan_local<<<(B_ * NC_ * D_) / 256, 256, 0, stream>>>(f_buf, i_buf, Pbuf, Abuf);
    k_scan_carry<<<(B_ * D_) / 256, 256, 0, stream>>>(Pbuf, Abuf, cybuf);
    k_scan_fix<<<(B_ * NC_ * D_) / 256, 256, 0, stream>>>(f_buf, i_buf, cybuf);

    k_ygate<<<M_, 256, 0, stream>>>(out, i_buf, gnw, yq, s_y);

    k_gemm_iu8<<<ggrid, 256, 0, stream>>>(yq, wq + 3 * (size_t)N_ * K_, out, s_y, s_w + 3, 0);
}
